// Cross_modalto_19267223290344
// MI455X (gfx1250) — matrix operand order repaired and checked by reading, not compiled, not run on hardware
//
#include <hip/hip_runtime.h>
#include <hip/hip_bf16.h>
#include <math.h>

// ---------------------------------------------------------------------------
// Problem constants (from reference): B=64, S=3600, D=128
// ---------------------------------------------------------------------------
#define BB 64
#define SS 3600
#define DD 128
#define MROWS (BB * SS)                  // 230400
#define NORM_FACT 0.17677669529663687f   // 1/sqrt(32)

typedef __attribute__((ext_vector_type(16))) __bf16 v16bf;
typedef __attribute__((ext_vector_type(8)))  float  v8f;
typedef __attribute__((ext_vector_type(4)))  unsigned int u32x4;
typedef __attribute__((ext_vector_type(4)))  int i32x4;
typedef __attribute__((ext_vector_type(8)))  int i32x8;

// ---------------------------------------------------------------------------
// TDM: DMA one 128x128 bf16 weight matrix (32 KB) from global into LDS at
// byte offset ldsOff. 1-D tensor descriptor per ISA 8.3/8.4. Caller must be
// wave-uniform; completion enforced with s_wait_tensorcnt.
// ---------------------------------------------------------------------------
static __device__ __forceinline__ void tdm_load_w(const __bf16* W, unsigned ldsOff) {
    const unsigned long long ga = (unsigned long long)(const void*)W;
    u32x4 g0;
    g0[0] = 1u;                                       // count=1, user D#
    g0[1] = ldsOff;                                   // lds_addr (bytes)
    g0[2] = (unsigned)(ga & 0xFFFFFFFFu);             // global_addr[31:0]
    g0[3] = (unsigned)((ga >> 32) & 0x1FFFFFFu)       // global_addr[56:32]
          | (2u << 30);                               // type = 2 ("image")
    i32x8 g1;
    g1[0] = (int)(1u << 16);                          // data_size=1 -> 2 bytes
    g1[1] = (int)((16384u & 0xFFFFu) << 16);          // tensor_dim0 = 16384 (b79:48)
    g1[2] = (int)(1u << 16);                          // tensor_dim1 = 1     (b111:80)
    g1[3] = (int)(16384u << 16);                      // tile_dim0 = 16384   (b127:112)
    g1[4] = 1;                                        // tile_dim1 = 1
    g1[5] = 16384;                                    // tensor_dim0_stride lo
    g1[6] = 0;
    g1[7] = 0;
    i32x4 z4 = (i32x4)0;
#if __clang_major__ >= 23
    i32x8 z8 = (i32x8)0;
    __builtin_amdgcn_tensor_load_to_lds(g0, g1, z4, z4, z8, 0);
#else
    __builtin_amdgcn_tensor_load_to_lds(g0, g1, z4, z4, 0);
#endif
    __builtin_amdgcn_s_wait_tensorcnt(0);
}

// ---------------------------------------------------------------------------
// Streaming f32 -> bf16 conversion (native casts -> hardware cvt ops).
// ---------------------------------------------------------------------------
__global__ __launch_bounds__(256) void f32_to_bf16(
    const float* __restrict__ in, __bf16* __restrict__ out, int n4)
{
    const int i = blockIdx.x * 256 + threadIdx.x;
    if (i >= n4) return;
    const float4 f = ((const float4*)in)[i];
    union { __bf16 h[4]; uint2 u; } r;
    r.h[0] = (__bf16)f.x;
    r.h[1] = (__bf16)f.y;
    r.h[2] = (__bf16)f.z;
    r.h[3] = (__bf16)f.w;
    ((uint2*)out)[i] = r.u;
}

// ---------------------------------------------------------------------------
// GEMM:  C[M x 128] = A[M x 128] @ W^T + bias   (A, W bf16; C f32)
// 128 threads = 4 waves; each wave computes a 16x128 strip (8 accum tiles).
// W staged into LDS via TDM; B fragments via ds_load_b128.
//   A frag (16x32 16-bit): row m0+rr, K chunk {16h..16h+15}
//   B frag (32x16 16-bit): col n (= W row n), K chunk {16h..16h+15}
// ---------------------------------------------------------------------------
__global__ __launch_bounds__(128) void gemm_bias_bf16(
    const __bf16* __restrict__ A, const __bf16* __restrict__ W,
    const float* __restrict__ bias, float* __restrict__ C, int M)
{
    extern __shared__ __bf16 sW[];             // 128*128 bf16 = 32 KB (dyn LDS)

    const int lane = threadIdx.x & 31;
    const int wave = threadIdx.x >> 5;

    if (wave == 0) tdm_load_w(W, 0u);
    __syncthreads();

    const int m0 = blockIdx.x * 64 + wave * 16;
    if (m0 >= M) return;                       // wave-uniform, after barrier

    const int half = lane >> 4;                // 0 or 1 (lane half)
    const int rr   = lane & 15;                // row (A) / col (B,C) in tile

    v8f acc[8];
#pragma unroll
    for (int i = 0; i < 8; ++i) acc[i] = (v8f)(0.0f);

    const __bf16* Arow = A + (size_t)(m0 + rr) * DD;

    union Frag { v16bf v; uint4 q[2]; };

#pragma unroll
    for (int ks = 0; ks < 4; ++ks) {
        const int kofs = ks * 32;

        Frag af;
        af.q[0] = *(const uint4*)(Arow + kofs + 16 * half);
        af.q[1] = *(const uint4*)(Arow + kofs + 16 * half + 8);

#pragma unroll
        for (int nt = 0; nt < 8; ++nt) {
            const __bf16* Wp = sW + (size_t)(nt * 16 + rr) * DD + kofs + 16 * half;
            Frag bf;
            bf.q[0] = *(const uint4*)(Wp);
            bf.q[1] = *(const uint4*)(Wp + 8);
            acc[nt] = __builtin_amdgcn_wmma_f32_16x16x32_bf16(
                false, af.v, false, bf.v, (short)0, acc[nt], false, false);
        }
    }

    // ---- store: C/D layout, VGPR r -> M = m0 + r + 8*half, N = nt*16 + rr
#pragma unroll
    for (int nt = 0; nt < 8; ++nt) {
        const int n    = nt * 16 + rr;
        const float bv = bias[n];
        const int mrow = m0 + 8 * half;
#pragma unroll
        for (int r = 0; r < 8; ++r) {
            C[(size_t)(mrow + r) * DD + n] = acc[nt][r] + bv;
        }
    }
}

// ---------------------------------------------------------------------------
// Fused dual GEMM: C0 = A@W0^T + b0, C1 = A@W1^T + b1 (K and V projections
// share the same A). Each A fragment feeds two WMMA chains. Both weight
// matrices staged into LDS by two concurrent TDM descriptors (waves 0 and 1).
// ---------------------------------------------------------------------------
__global__ __launch_bounds__(128) void gemm_bias_bf16_dual(
    const __bf16* __restrict__ A,
    const __bf16* __restrict__ W0, const __bf16* __restrict__ W1,
    const float* __restrict__ bias0, const float* __restrict__ bias1,
    float* __restrict__ C0, float* __restrict__ C1, int M)
{
    extern __shared__ __bf16 sW2[];            // 2 * 32 KB (dyn LDS)

    const int lane = threadIdx.x & 31;
    const int wave = threadIdx.x >> 5;

    if (wave == 0)      tdm_load_w(W0, 0u);
    else if (wave == 1) tdm_load_w(W1, 32768u);
    __syncthreads();

    const int m0 = blockIdx.x * 64 + wave * 16;
    if (m0 >= M) return;                       // wave-uniform, after barrier

    const int half = lane >> 4;
    const int rr   = lane & 15;

    v8f acc0[8], acc1[8];
#pragma unroll
    for (int i = 0; i < 8; ++i) { acc0[i] = (v8f)(0.0f); acc1[i] = (v8f)(0.0f); }

    const __bf16* Arow = A + (size_t)(m0 + rr) * DD;
    const __bf16* sB0  = sW2;
    const __bf16* sB1  = sW2 + 16384;

    union Frag { v16bf v; uint4 q[2]; };

#pragma unroll
    for (int ks = 0; ks < 4; ++ks) {
        const int kofs = ks * 32;

        Frag af;
        af.q[0] = *(const uint4*)(Arow + kofs + 16 * half);
        af.q[1] = *(const uint4*)(Arow + kofs + 16 * half + 8);

#pragma unroll
        for (int nt = 0; nt < 8; ++nt) {
            const size_t wo = (size_t)(nt * 16 + rr) * DD + kofs + 16 * half;
            Frag b0, b1;
            b0.q[0] = *(const uint4*)(sB0 + wo);
            b0.q[1] = *(const uint4*)(sB0 + wo + 8);
            b1.q[0] = *(const uint4*)(sB1 + wo);
            b1.q[1] = *(const uint4*)(sB1 + wo + 8);
            acc0[nt] = __builtin_amdgcn_wmma_f32_16x16x32_bf16(
                false, af.v, false, b0.v, (short)0, acc0[nt], false, false);
            acc1[nt] = __builtin_amdgcn_wmma_f32_16x16x32_bf16(
                false, af.v, false, b1.v, (short)0, acc1[nt], false, false);
        }
    }

#pragma unroll
    for (int nt = 0; nt < 8; ++nt) {
        const int n    = nt * 16 + rr;
        const float b0 = bias0[n];
        const float b1 = bias1[n];
        const int mrow = m0 + 8 * half;
#pragma unroll
        for (int r = 0; r < 8; ++r) {
            C0[(size_t)(mrow + r) * DD + n] = acc0[nt][r] + b0;
            C1[(size_t)(mrow + r) * DD + n] = acc1[nt][r] + b1;
        }
    }
}

// ---------------------------------------------------------------------------
// Cross-covariance attention: one block per (h,b2) in [16)x[64).
// Q/K/V slices are contiguous 3600x8 f32 chunks at base=(h*230400+b2*3600)*8.
// scores[8][8] = sum_s l2n(Qrow) x l2n(Krow), softmax rows, att = probs*V^T.
// Output relayout (written directly as bf16 for the Wo GEMM):
//   Ar[((h*3600+s)*64+b2)*8 + c]
// ---------------------------------------------------------------------------
__global__ __launch_bounds__(256) void xcov_attn(
    const float* __restrict__ Qb, const float* __restrict__ Kb,
    const float* __restrict__ Vb, __bf16* __restrict__ Ar)
{
    const int h  = blockIdx.x >> 6;
    const int b2 = blockIdx.x & 63;
    const size_t base = ((size_t)h * (size_t)(64 * SS) + (size_t)b2 * SS) * 8;
    const int t = threadIdx.x;

    __shared__ float s_p[64];
    if (t < 64) s_p[t] = 0.0f;
    __syncthreads();

    float acc[8][8];
#pragma unroll
    for (int c = 0; c < 8; ++c)
#pragma unroll
        for (int d = 0; d < 8; ++d) acc[c][d] = 0.0f;

    for (int s2 = t; s2 < SS; s2 += 256) {
        const float4* qp = (const float4*)(Qb + base + (size_t)s2 * 8);
        const float4* kp = (const float4*)(Kb + base + (size_t)s2 * 8);
        __builtin_prefetch(qp + 512, 0, 0);      // (s2 + 256) row
        __builtin_prefetch(kp + 512, 0, 0);
        float4 q0 = qp[0], q1 = qp[1];
        float4 k0 = kp[0], k1 = kp[1];
        float qv[8] = {q0.x, q0.y, q0.z, q0.w, q1.x, q1.y, q1.z, q1.w};
        float kv[8] = {k0.x, k0.y, k0.z, k0.w, k1.x, k1.y, k1.z, k1.w};
        float qs = 0.f, ks = 0.f;
#pragma unroll
        for (int i = 0; i < 8; ++i) { qs += qv[i] * qv[i]; ks += kv[i] * kv[i]; }
        const float sc = (1.0f / fmaxf(sqrtf(qs), 1e-12f)) *
                         (1.0f / fmaxf(sqrtf(ks), 1e-12f));
#pragma unroll
        for (int c = 0; c < 8; ++c)
#pragma unroll
            for (int d = 0; d < 8; ++d) acc[c][d] += (qv[c] * kv[d]) * sc;
    }

    // wave reduction, then cross-wave via LDS float atomics
#pragma unroll
    for (int c = 0; c < 8; ++c)
#pragma unroll
        for (int d = 0; d < 8; ++d) {
            float x = acc[c][d];
            x += __shfl_xor(x, 16, 32);
            x += __shfl_xor(x, 8, 32);
            x += __shfl_xor(x, 4, 32);
            x += __shfl_xor(x, 2, 32);
            x += __shfl_xor(x, 1, 32);
            if ((t & 31) == 0) atomicAdd(&s_p[c * 8 + d], x);
        }
    __syncthreads();

    // softmax over last dim (rows of 8), threads 0..7
    if (t < 8) {
        float row[8], mx = -3.0e38f;
#pragma unroll
        for (int d = 0; d < 8; ++d) {
            row[d] = s_p[t * 8 + d] * NORM_FACT;
            mx = fmaxf(mx, row[d]);
        }
        float sum = 0.f;
#pragma unroll
        for (int d = 0; d < 8; ++d) { row[d] = expf(row[d] - mx); sum += row[d]; }
        const float inv = 1.0f / sum;
#pragma unroll
        for (int d = 0; d < 8; ++d) s_p[t * 8 + d] = row[d] * inv;
    }
    __syncthreads();

    float p[8][8];
#pragma unroll
    for (int c = 0; c < 8; ++c)
#pragma unroll
        for (int d = 0; d < 8; ++d) p[c][d] = s_p[c * 8 + d];

    for (int s2 = t; s2 < SS; s2 += 256) {
        const float4* vp = (const float4*)(Vb + base + (size_t)s2 * 8);
        __builtin_prefetch(vp + 512, 0, 0);
        float4 v0 = vp[0], v1 = vp[1];
        float vv[8] = {v0.x, v0.y, v0.z, v0.w, v1.x, v1.y, v1.z, v1.w};
        union { __bf16 h[8]; uint4 u; } r;
#pragma unroll
        for (int c = 0; c < 8; ++c) {
            float s = 0.f;
#pragma unroll
            for (int d = 0; d < 8; ++d) s += p[c][d] * vv[d];
            r.h[c] = (__bf16)s;
        }
        __bf16* op = Ar + (((size_t)h * SS + s2) * 64 + b2) * 8;
        *(uint4*)op = r.u;
    }
}

// ---------------------------------------------------------------------------
// z = LayerNorm(v + sub) * g + b  + v     (one wave per 128-wide row)
// ---------------------------------------------------------------------------
__global__ __launch_bounds__(256) void add_norm_res(
    const float* __restrict__ Vin, const float* __restrict__ Sub,
    const float* __restrict__ g, const float* __restrict__ bta,
    float* __restrict__ Z)
{
    const int row  = blockIdx.x * 8 + (threadIdx.x >> 5);
    const int lane = threadIdx.x & 31;
    const size_t off = (size_t)row * DD + lane * 4;

    float4 vv = *(const float4*)(Vin + off);
    float4 ss = *(const float4*)(Sub + off);
    float x0 = vv.x + ss.x, x1 = vv.y + ss.y, x2 = vv.z + ss.z, x3 = vv.w + ss.w;

    float s  = x0 + x1 + x2 + x3;
    float sq = x0 * x0 + x1 * x1 + x2 * x2 + x3 * x3;
#pragma unroll
    for (int o = 16; o >= 1; o >>= 1) {
        s  += __shfl_xor(s,  o, 32);
        sq += __shfl_xor(sq, o, 32);
    }
    const float mean = s * (1.0f / 128.0f);
    const float var  = sq * (1.0f / 128.0f) - mean * mean;
    const float rstd = rsqrtf(var + 1e-5f);

    float4 gg = *(const float4*)(g + lane * 4);
    float4 bb = *(const float4*)(bta + lane * 4);
    float4 out;
    out.x = (x0 - mean) * rstd * gg.x + bb.x + vv.x;
    out.y = (x1 - mean) * rstd * gg.y + bb.y + vv.y;
    out.z = (x2 - mean) * rstd * gg.z + bb.z + vv.z;
    out.w = (x3 - mean) * rstd * gg.w + bb.w + vv.w;
    *(float4*)(Z + off) = out;
}

// ---------------------------------------------------------------------------
// Conv2d (1->16, kernel 15x16, stride 15x16, VALID) + exact GELU + flatten.
// One thread per (b, i, j); computes all 16 filters; weights staged in LDS.
// out[b*30720 + f*1920 + i*8 + j]
// ---------------------------------------------------------------------------
__global__ __launch_bounds__(256) void conv_gelu(
    const float* __restrict__ Zb, const float* __restrict__ Wc,
    float* __restrict__ Out)
{
    __shared__ float sw[16 * 240];
    for (int i = threadIdx.x; i < 16 * 240; i += 256) sw[i] = Wc[i];
    __syncthreads();

    const int idx = blockIdx.x * 256 + threadIdx.x;   // over 64*240*8 = 122880
    if (idx >= BB * 240 * 8) return;
    const int j = idx & 7;
    const int i = (idx >> 3) % 240;
    const int b = idx / (240 * 8);

    float fa[16];
#pragma unroll
    for (int f = 0; f < 16; ++f) fa[f] = 0.0f;

    const float* zp = Zb + ((size_t)b * SS + (size_t)i * 15) * DD + j * 16;
    for (int r = 0; r < 15; ++r) {
        __builtin_prefetch(zp + (size_t)(r + 1) * DD, 0, 3);
        const float4* rp = (const float4*)(zp + (size_t)r * DD);
        float4 a0 = rp[0], a1 = rp[1], a2 = rp[2], a3 = rp[3];
        float xr[16] = {a0.x, a0.y, a0.z, a0.w, a1.x, a1.y, a1.z, a1.w,
                        a2.x, a2.y, a2.z, a2.w, a3.x, a3.y, a3.z, a3.w};
#pragma unroll
        for (int f = 0; f < 16; ++f) {
            const float* wrow = &sw[f * 240 + r * 16];
#pragma unroll
            for (int tt = 0; tt < 16; ++tt) fa[f] += xr[tt] * wrow[tt];
        }
    }
#pragma unroll
    for (int f = 0; f < 16; ++f) {
        const float x = fa[f];
        const float gv = 0.5f * x * (1.0f + erff(x * 0.7071067811865475f));
        Out[(size_t)b * 30720 + (size_t)f * 1920 + i * 8 + j] = gv;
    }
}

// ---------------------------------------------------------------------------
// Launches. Workspace layout (bytes):
//   Qb f32[MD] | Kb f32[MD] | Vb f32[MD] | vbf bf16[MD] | qbf bf16[MD]
//   | Ar bf16[MD] | Wq/Wk/Wv/Wo bf16[16384] each
// sub reuses Qb, z reuses Kb. (Ar must be distinct: relayout crosses blocks.)
// ---------------------------------------------------------------------------
extern "C" void kernel_launch(void* const* d_in, const int* in_sizes, int n_in,
                              void* d_out, int out_size, void* d_ws, size_t ws_size,
                              hipStream_t stream) {
    (void)in_sizes; (void)n_in; (void)out_size; (void)ws_size;

    const float* q    = (const float*)d_in[0];
    const float* v    = (const float*)d_in[1];
    const float* Wq   = (const float*)d_in[2];
    const float* bq   = (const float*)d_in[3];
    const float* Wk   = (const float*)d_in[4];
    const float* bk   = (const float*)d_in[5];
    const float* Wv   = (const float*)d_in[6];
    const float* bv   = (const float*)d_in[7];
    const float* Wo   = (const float*)d_in[8];
    const float* bo   = (const float*)d_in[9];
    const float* ln_g = (const float*)d_in[10];
    const float* ln_b = (const float*)d_in[11];
    const float* cw   = (const float*)d_in[12];
    float* out = (float*)d_out;

    const size_t MD = (size_t)MROWS * DD;     // 29,491,200 elements
    char* ws = (char*)d_ws;
    float*  Qb  = (float*)(ws);                            // -> later: sub
    float*  Kb  = (float*)(ws + 4 * MD);                   // -> later: z
    float*  Vb  = (float*)(ws + 8 * MD);
    __bf16* vbf = (__bf16*)(ws + 12 * MD);
    __bf16* qbf = (__bf16*)(ws + 14 * MD);
    __bf16* Arb = (__bf16*)(ws + 16 * MD);
    __bf16* Wqb = (__bf16*)(ws + 18 * MD);
    __bf16* Wkb = Wqb + (size_t)DD * DD;
    __bf16* Wvb = Wkb + (size_t)DD * DD;
    __bf16* Wob = Wvb + (size_t)DD * DD;

    const int cvtBlocks  = (int)(MD / 4 / 256);            // 28800
    const int wBlocks    = (DD * DD / 4 + 255) / 256;      // 16
    const int gemmBlocks = MROWS / 64;                     // 3600
    const size_t ldsW    = (size_t)DD * DD * sizeof(__bf16);   // 32 KB
    const size_t ldsW2   = 2 * ldsW;                           // 64 KB

    // pre-convert activations + weights to bf16
    f32_to_bf16<<<cvtBlocks, 256, 0, stream>>>(v,  vbf, (int)(MD / 4));
    f32_to_bf16<<<cvtBlocks, 256, 0, stream>>>(q,  qbf, (int)(MD / 4));
    f32_to_bf16<<<wBlocks,   256, 0, stream>>>(Wq, Wqb, DD * DD / 4);
    f32_to_bf16<<<wBlocks,   256, 0, stream>>>(Wk, Wkb, DD * DD / 4);
    f32_to_bf16<<<wBlocks,   256, 0, stream>>>(Wv, Wvb, DD * DD / 4);
    f32_to_bf16<<<wBlocks,   256, 0, stream>>>(Wo, Wob, DD * DD / 4);

    // Q = v @ Wq^T + bq
    gemm_bias_bf16<<<gemmBlocks, 128, ldsW, stream>>>(vbf, Wqb, bq, Qb, MROWS);
    // K = q @ Wk^T + bk ; V = q @ Wv^T + bv  (fused: shared A fragments)
    gemm_bias_bf16_dual<<<gemmBlocks, 128, ldsW2, stream>>>(
        qbf, Wkb, Wvb, bk, bv, Kb, Vb, MROWS);

    // per-(h,b2) cross-covariance attention -> Arb (bf16, relayouted)
    xcov_attn<<<16 * 64, 256, 0, stream>>>(Qb, Kb, Vb, Arb);

    // sub = Ar @ Wo^T + bo   (into Qb, now dead)
    gemm_bias_bf16<<<gemmBlocks, 128, ldsW, stream>>>(Arb, Wob, bo, Qb, MROWS);

    // z = LN(v + sub)*g + b + v   (into Kb, now dead)
    add_norm_res<<<MROWS / 8, 256, 0, stream>>>(v, Qb, ln_g, ln_b, Kb);

    // conv(15x16 stride 15x16) + GELU + flatten
    conv_gelu<<<(BB * 240 * 8) / 256, 256, 0, stream>>>(Kb, cw, out);
}
